// RWKVFeedForward_45337674777326
// MI455X (gfx1250) — matrix operand order repaired and checked by reading, not compiled, not run on hardware
//
#include <hip/hip_runtime.h>
#include <hip/hip_bf16.h>

typedef __attribute__((ext_vector_type(16))) _Float16 v16h;
typedef __attribute__((ext_vector_type(8)))  float    v8f;

#define DIM    2048
#define HIDDEN 2816
#define BATCH  4
#define SEQ    2048
#define BT     (BATCH * SEQ)          // 8192 rows
#define LN_EPS 1e-5f

#define LDS_STRIDE 40                 // 32 halfs data + 8 halfs pad (80B rows: bank-conflict-free)

// ---------------------------------------------------------------------------
// f32 -> f16 conversion (weights)
// ---------------------------------------------------------------------------
__global__ __launch_bounds__(256) void cvt_f32_f16_kernel(const float* __restrict__ src,
                                                          _Float16* __restrict__ dst,
                                                          long long n) {
    long long i = (long long)blockIdx.x * blockDim.x + threadIdx.x;
    long long stride = (long long)gridDim.x * blockDim.x;
    for (; i < n; i += stride) dst[i] = (_Float16)src[i];
}

// ---------------------------------------------------------------------------
// Token-shift premix: xk = x + (x_prev - x)*tmk ; xr = x + (x_prev - x)*tmr
// Position 0 mixes with position 1 (faithful to reference).
// ---------------------------------------------------------------------------
__global__ __launch_bounds__(256) void premix_kernel(const float* __restrict__ x,
                                                     const float* __restrict__ tmk,
                                                     const float* __restrict__ tmr,
                                                     _Float16* __restrict__ xk,
                                                     _Float16* __restrict__ xr) {
    long long idx = (long long)blockIdx.x * blockDim.x + threadIdx.x;
    const long long total = (long long)BT * DIM;
    if (idx >= total) return;
    int c = (int)(idx % DIM);
    long long bt = idx / DIM;
    int t = (int)(bt % SEQ);
    long long prev = (t == 0) ? (idx + DIM) : (idx - DIM);
    float xv = x[idx];
    float xx = x[prev] - xv;
    xk[idx] = (_Float16)(xv + xx * tmk[c]);
    xr[idx] = (_Float16)(xv + xx * tmr[c]);
}

// ---------------------------------------------------------------------------
// WMMA fragment loads from padded LDS tiles (row stride = LDS_STRIDE halfs).
// A layout (16-bit 16x32, ISA 7.12.2): lane m=L&15, half=L>>4 holds
//   K = [8*half, 8*half+8) then [16+8*half, +8)  -> 2x ds_load_b128.
// B layout (dense 32x16): lane n, half holds K = [8*half, +8) then [16+8*half, +8) (same as A).
// ---------------------------------------------------------------------------
__device__ __forceinline__ v16h load_fragA(const _Float16* tile, int row0, int lane) {
    int m = lane & 15, half = lane >> 4;
    const _Float16* p = tile + (row0 + m) * LDS_STRIDE;
    union { v16h h; uint4 q[2]; } f;
    f.q[0] = *(const uint4*)(p + (half << 3));
    f.q[1] = *(const uint4*)(p + 16 + (half << 3));
    return f.h;
}

__device__ __forceinline__ v16h load_fragB(const _Float16* tile, int col0, int lane) {
    int n = lane & 15, half = lane >> 4;
    const _Float16* p = tile + (col0 + n) * LDS_STRIDE;
    union { v16h h; uint4 q[2]; } f;
    f.q[0] = *(const uint4*)(p + (half << 3));
    f.q[1] = *(const uint4*)(p + 16 + (half << 3));
    return f.h;
}

// ---------------------------------------------------------------------------
// Tiled NT GEMM:  out[M,N] = A[M,K] * B[N,K]^T   (A,B f16 row-major, acc f32)
// Block tile 256x128, 8 waves arranged 4(M) x 2(N); wave tile 64x64 (4x4 WMMA).
// Double-buffered LDS, one barrier per K-step, K-step = 32.
// M multiple of 256, N multiple of 128, K multiple of 32 (true for all GEMMs).
// ---------------------------------------------------------------------------
template <bool HALF_OUT>
__global__ __launch_bounds__(256) void gemm_nt_wmma(const _Float16* __restrict__ A,
                                                    const _Float16* __restrict__ Bw,
                                                    void* __restrict__ out,
                                                    int M, int N, int K) {
    __shared__ __align__(16) _Float16 As[2][256 * LDS_STRIDE];   // 2 x 20 KB
    __shared__ __align__(16) _Float16 Bs[2][128 * LDS_STRIDE];   // 2 x 10 KB

    const int tid  = threadIdx.x;
    const int lane = tid & 31;
    const int wave = tid >> 5;
    const int wm   = wave & 3;   // 0..3 -> 64-row M sub-tile
    const int wn   = wave >> 2;  // 0..1 -> 64-col N sub-tile
    const int blockRow = blockIdx.y * 256;
    const int blockCol = blockIdx.x * 128;

    v8f acc[4][4];
#pragma unroll
    for (int i = 0; i < 4; ++i)
#pragma unroll
        for (int j = 0; j < 4; ++j)
#pragma unroll
            for (int v = 0; v < 8; ++v) acc[i][j][v] = 0.0f;

    // cooperative tile loads:
    //   A: 256 rows -> 1 row per thread, 32 halfs = 4x uint4
    //   B: 128 rows -> 2 threads per row, 16 halfs = 2x uint4
    const int rowA = tid;
    const int rowB = tid >> 1;
    const int segB = (tid & 1) << 4;
    const uint4* gA = (const uint4*)(A  + (long long)(blockRow + rowA) * K);
    const uint4* gB = (const uint4*)(Bw + (long long)(blockCol + rowB) * K + segB);

    // ---- prologue: tile 0 into buffer 0
    {
        uint4 ra0 = gA[0], ra1 = gA[1], ra2 = gA[2], ra3 = gA[3];
        uint4 rb0 = gB[0], rb1 = gB[1];
        gA += 4; gB += 4;
        uint4* sA = (uint4*)(&As[0][rowA * LDS_STRIDE]);
        sA[0] = ra0; sA[1] = ra1; sA[2] = ra2; sA[3] = ra3;
        uint4* sB = (uint4*)(&Bs[0][rowB * LDS_STRIDE + segB]);
        sB[0] = rb0; sB[1] = rb1;
    }

    int buf = 0;
    for (int k0 = 32; k0 < K; k0 += 32) {
        __syncthreads();
        // issue next tile's global loads before compute (latency overlap)
        uint4 ra0 = gA[0], ra1 = gA[1], ra2 = gA[2], ra3 = gA[3];
        uint4 rb0 = gB[0], rb1 = gB[1];
        __builtin_prefetch((const void*)(gA + 4), 0, 3);   // warm next-next tile
        __builtin_prefetch((const void*)(gB + 4), 0, 3);
        gA += 4; gB += 4;

        // compute on current buffer
        {
            const _Float16* At = As[buf];
            const _Float16* Bt = Bs[buf];
            v16h af[4];
#pragma unroll
            for (int i = 0; i < 4; ++i) af[i] = load_fragA(At, wm * 64 + i * 16, lane);
#pragma unroll
            for (int j = 0; j < 4; ++j) {
                v16h bf = load_fragB(Bt, wn * 64 + j * 16, lane);
#pragma unroll
                for (int i = 0; i < 4; ++i)
                    acc[i][j] = __builtin_amdgcn_wmma_f32_16x16x32_f16(
                        false, af[i], false, bf, (short)0, acc[i][j], false, false);
            }
        }

        // stage next tile into the other buffer
        uint4* sA = (uint4*)(&As[buf ^ 1][rowA * LDS_STRIDE]);
        sA[0] = ra0; sA[1] = ra1; sA[2] = ra2; sA[3] = ra3;
        uint4* sB = (uint4*)(&Bs[buf ^ 1][rowB * LDS_STRIDE + segB]);
        sB[0] = rb0; sB[1] = rb1;
        buf ^= 1;
    }

    // ---- epilogue: last tile
    __syncthreads();
    {
        const _Float16* At = As[buf];
        const _Float16* Bt = Bs[buf];
        v16h af[4];
#pragma unroll
        for (int i = 0; i < 4; ++i) af[i] = load_fragA(At, wm * 64 + i * 16, lane);
#pragma unroll
        for (int j = 0; j < 4; ++j) {
            v16h bf = load_fragB(Bt, wn * 64 + j * 16, lane);
#pragma unroll
            for (int i = 0; i < 4; ++i)
                acc[i][j] = __builtin_amdgcn_wmma_f32_16x16x32_f16(
                    false, af[i], false, bf, (short)0, acc[i][j], false, false);
        }
    }

    // D layout (f32 16x16): lane n=L&15; VGPR v -> row v + 8*(L>>4)
    const int hrow = (lane >> 4) << 3;
    const int ncol = lane & 15;
#pragma unroll
    for (int i = 0; i < 4; ++i)
#pragma unroll
        for (int j = 0; j < 4; ++j) {
            int r0 = blockRow + wm * 64 + i * 16 + hrow;
            int c0 = blockCol + wn * 64 + j * 16 + ncol;
#pragma unroll
            for (int v = 0; v < 8; ++v) {
                long long o = (long long)(r0 + v) * N + c0;
                if (HALF_OUT) ((_Float16*)out)[o] = (_Float16)acc[i][j][v];
                else          ((float*)out)[o]    = acc[i][j][v];
            }
        }
}

// ---------------------------------------------------------------------------
// Sequential decay scan over T: state = state*w + relu(k); a = state*r
// One thread per (b,h) channel; f32 state, f16 i/o.
// ---------------------------------------------------------------------------
__global__ __launch_bounds__(256) void scan_kernel(const _Float16* __restrict__ kh,
                                                   const _Float16* __restrict__ rh,
                                                   const float* __restrict__ td,
                                                   _Float16* __restrict__ ah) {
    int idx = blockIdx.x * blockDim.x + threadIdx.x;
    if (idx >= BATCH * HIDDEN) return;
    int b = idx / HIDDEN;
    int h = idx % HIDDEN;
    float w = __expf(-__expf(td[h]));
    float state = 0.0f;
    long long base = (long long)b * SEQ * HIDDEN + h;
#pragma unroll 4
    for (int t = 0; t < SEQ; ++t) {
        long long o = base + (long long)t * HIDDEN;
        float kv = (float)kh[o];
        float rv = (float)rh[o];
        state = state * w + fmaxf(kv, 0.0f);
        ah[o] = (_Float16)(state * rv);
    }
}

// ---------------------------------------------------------------------------
// In-place LayerNorm over last dim (C=2048), one block per row.
// ---------------------------------------------------------------------------
__global__ __launch_bounds__(256) void layernorm_kernel(float* __restrict__ out,
                                                        const float* __restrict__ gamma,
                                                        const float* __restrict__ beta) {
    __shared__ float s1[256];
    __shared__ float s2[256];
    float* p = out + (long long)blockIdx.x * DIM;
    float a = 0.0f, b = 0.0f;
    for (int c = threadIdx.x; c < DIM; c += 256) {
        float v = p[c];
        a += v; b += v * v;
    }
    s1[threadIdx.x] = a; s2[threadIdx.x] = b;
    __syncthreads();
    for (int off = 128; off > 0; off >>= 1) {
        if (threadIdx.x < off) {
            s1[threadIdx.x] += s1[threadIdx.x + off];
            s2[threadIdx.x] += s2[threadIdx.x + off];
        }
        __syncthreads();
    }
    float mu  = s1[0] * (1.0f / DIM);
    float var = s2[0] * (1.0f / DIM) - mu * mu;
    float rs  = rsqrtf(var + LN_EPS);
    for (int c = threadIdx.x; c < DIM; c += 256) {
        p[c] = (p[c] - mu) * rs * gamma[c] + beta[c];
    }
}

// ---------------------------------------------------------------------------
// Host launcher
// ---------------------------------------------------------------------------
extern "C" void kernel_launch(void* const* d_in, const int* in_sizes, int n_in,
                              void* d_out, int out_size, void* d_ws, size_t ws_size,
                              hipStream_t stream) {
    const float* x    = (const float*)d_in[0];
    const float* tmk  = (const float*)d_in[1];
    const float* tmr  = (const float*)d_in[2];
    const float* Wk   = (const float*)d_in[3];
    const float* Wr   = (const float*)d_in[4];
    const float* Wv   = (const float*)d_in[5];
    const float* td   = (const float*)d_in[6];
    const float* gam  = (const float*)d_in[7];
    const float* bet  = (const float*)d_in[8];
    float* out = (float*)d_out;

    // workspace layout (all region sizes are multiples of 256B)
    const long long W_ELEMS  = (long long)HIDDEN * DIM;   // 5,767,168
    const long long BT_C     = (long long)BT * DIM;       // 16,777,216
    const long long BT_H     = (long long)BT * HIDDEN;    // 23,068,672
    char* ws = (char*)d_ws;
    _Float16* Wkh = (_Float16*)(ws);
    _Float16* Wrh = Wkh + W_ELEMS;
    _Float16* Wvh = Wrh + W_ELEMS;
    _Float16* kh  = Wvh + W_ELEMS;
    _Float16* rh  = kh  + BT_H;
    _Float16* xkh = rh  + BT_H;
    _Float16* xrh = xkh + BT_C;
    _Float16* ah  = xkh;            // reuse xk region once GEMMs 1&2 are done

    // 1) weight conversion f32 -> f16
    cvt_f32_f16_kernel<<<4096, 256, 0, stream>>>(Wk, Wkh, W_ELEMS);
    cvt_f32_f16_kernel<<<4096, 256, 0, stream>>>(Wr, Wrh, W_ELEMS);
    cvt_f32_f16_kernel<<<4096, 256, 0, stream>>>(Wv, Wvh, W_ELEMS);

    // 2) token-shift premix -> xk, xr (f16)
    premix_kernel<<<(unsigned)((BT_C + 255) / 256), 256, 0, stream>>>(x, tmk, tmr, xkh, xrh);

    // 3) k = xk @ Wk^T, r = xr @ Wr^T   [8192 x 2816], K = 2048
    {
        dim3 grid(HIDDEN / 128, BT / 256);
        gemm_nt_wmma<true><<<grid, 256, 0, stream>>>(xkh, Wkh, (void*)kh, BT, HIDDEN, DIM);
        gemm_nt_wmma<true><<<grid, 256, 0, stream>>>(xrh, Wrh, (void*)rh, BT, HIDDEN, DIM);
    }

    // 4) sequential decay scan -> a (f16)
    scan_kernel<<<(BATCH * HIDDEN + 255) / 256, 256, 0, stream>>>(kh, rh, td, ah);

    // 5) out = a @ Wv^T  [8192 x 2048], K = 2816  (f32 straight into d_out)
    {
        dim3 grid(DIM / 128, BT / 256);
        gemm_nt_wmma<false><<<grid, 256, 0, stream>>>(ah, Wvh, (void*)out, BT, DIM, HIDDEN);
    }

    // 6) in-place LayerNorm over C
    layernorm_kernel<<<BT, 256, 0, stream>>>(out, gam, bet);
}
